// MaskedAttentionHead_3547642987204
// MI455X (gfx1250) — matrix operand order repaired and checked by reading, not compiled, not run on hardware
//
#include <hip/hip_runtime.h>
#include <hip/hip_bf16.h>

// ---------------------------------------------------------------------------
// MaskedAttentionHead for MI455X (gfx1250, wave32, WMMA 16x16x32 bf16).
// Split-precision bf16 (hi/lo) emulated-fp32 GEMMs + flash-attention.
// B=4, S=2048, E=1024, H=128. Workspace: ~13.6 MB.
// ---------------------------------------------------------------------------

typedef __attribute__((ext_vector_type(16))) __bf16 v16bf;
typedef __attribute__((ext_vector_type(8)))  __bf16 v8bf;
typedef __attribute__((ext_vector_type(8)))  float  v8f;
typedef __attribute__((ext_vector_type(4)))  float  v4f;

#define WMMA_BF16(A, B, C) \
  __builtin_amdgcn_wmma_f32_16x16x32_bf16(false, (A), false, (B), (short)0, (C), false, false)

static constexpr int Bb = 4, S = 2048, E = 1024, H = 128;
static constexpr int BS = Bb * S;  // 8192 total rows

struct bf2 { __bf16 hi, lo; };
static __device__ inline bf2 splitf(float f) {
  __bf16 h = (__bf16)f;
  return { h, (__bf16)(f - (float)h) };
}
static __device__ inline v16bf cat16(v8bf a, v8bf b) {
  v16bf r;
#pragma unroll
  for (int i = 0; i < 8; ++i) { r[i] = a[i]; r[i + 8] = b[i]; }
  return r;
}

// ---------------------------------------------------------------------------
// Kernel 0: one-time conversion of Wq/Wk/Wv (f32) -> concatenated bf16 hi/lo
// planes [3][H][E].  393216 elements, 8 per thread, 192 blocks x 256.
// ---------------------------------------------------------------------------
__global__ __launch_bounds__(256) void conv_w_kernel(
    const float* __restrict__ Wq, const float* __restrict__ Wk,
    const float* __restrict__ Wv,
    __bf16* __restrict__ whi, __bf16* __restrict__ wlo)
{
  const size_t e0 = ((size_t)blockIdx.x * blockDim.x + threadIdx.x) * 8;
  const size_t HE = (size_t)H * E;
  const int proj = (int)(e0 / HE);
  const size_t off = e0 % HE;
  const float* src = ((proj == 0) ? Wq : (proj == 1) ? Wk : Wv) + off;
  v4f f0 = *(const v4f*)(src);
  v4f f1 = *(const v4f*)(src + 4);
  v8bf h, l;
#pragma unroll
  for (int i = 0; i < 4; ++i) {
    bf2 a = splitf(f0[i]); h[i] = a.hi;     l[i] = a.lo;
    bf2 b = splitf(f1[i]); h[i + 4] = b.hi; l[i + 4] = b.lo;
  }
  *(v8bf*)(whi + e0) = h;
  *(v8bf*)(wlo + e0) = l;
}

// ---------------------------------------------------------------------------
// Kernel 1: Q/K/V projections.  One wave = (proj, 16-row tile of B*S).
// Waves in a block share the same x rows across the 3 projections (WGP$ reuse).
// A-layout (16x32 bf16):  lane<16: m=lane, K {0..15};       lane>=16: K {16..31}
// B-layout (32x16 bf16):  lane<16: n=lane, K {0..15};       lane>=16: K {16..31}
// C-layout (16x16 f32):   VGPR r: M=r (lanes 0-15) / M=r+8 (lanes 16-31), N=lane%16
// ---------------------------------------------------------------------------
__global__ __launch_bounds__(256) void qkv_proj_kernel(
    const float* __restrict__ x,
    const __bf16* __restrict__ whi, const __bf16* __restrict__ wlo,
    __bf16* __restrict__ qhi, __bf16* __restrict__ qlo,
    __bf16* __restrict__ khi, __bf16* __restrict__ klo,
    __bf16* __restrict__ vthi, __bf16* __restrict__ vtlo)
{
  const int lane = threadIdx.x & 31;
  const int wid  = blockIdx.x * (blockDim.x >> 5) + (threadIdx.x >> 5);  // 0..1535
  const int proj = wid % 3;        // adjacent waves share the same mtile
  const int mt   = wid / 3;        // 16-row tile of B*S

  const int  n       = lane & 15;
  const bool hiHalf  = lane >= 16;
  const int  akb     = hiHalf ? 16 : 0;   // A-fragment k chunk base
  const int  bkb     = hiHalf ? 16 : 0;   // B-fragment k chunk base
  const int  half8   = hiHalf ? 8  : 0;
  const int  arow    = mt * 16 + n;       // x row this lane supplies (m = lane%16)
  const size_t wrow0 = (size_t)proj * H;  // first row of this projection's W

  v8f acc[8] = {};  // 8 N-tiles of H (128 = 8*16)

  for (int k0 = 0; k0 < E; k0 += 32) {
    // A fragment from x (fp32 -> bf16 hi/lo). One contiguous 16-float chunk.
    const float* ap = x + (size_t)arow * E + k0 + akb;
    v4f a0 = *(const v4f*)(ap);      v4f a1 = *(const v4f*)(ap + 4);
    v4f a2 = *(const v4f*)(ap + 8); v4f a3 = *(const v4f*)(ap + 12);
    float af[16];
#pragma unroll
    for (int i = 0; i < 4; ++i) { af[i] = a0[i]; af[4+i] = a1[i]; af[8+i] = a2[i]; af[12+i] = a3[i]; }
    v16bf ahi, alo;
#pragma unroll
    for (int i = 0; i < 16; ++i) { bf2 t = splitf(af[i]); ahi[i] = t.hi; alo[i] = t.lo; }

#pragma unroll
    for (int t = 0; t < 8; ++t) {
      // B fragment straight from precomputed bf16 planes: 16 contiguous bf16.
      const size_t wo = (wrow0 + t * 16 + n) * E + k0 + bkb;
      v16bf bhi = *(const v16bf*)(whi + wo);
      v16bf blo = *(const v16bf*)(wlo + wo);
      // 3-term split-precision product: hi*hi + hi*lo + lo*hi  (~fp32 accuracy)
      acc[t] = WMMA_BF16(ahi, bhi, acc[t]);
      acc[t] = WMMA_BF16(ahi, blo, acc[t]);
      acc[t] = WMMA_BF16(alo, bhi, acc[t]);
    }
  }

  // Store results as bf16 hi/lo planes. Q,K row-major [BS][H]; V transposed [H][BS].
#pragma unroll
  for (int t = 0; t < 8; ++t) {
#pragma unroll
    for (int r = 0; r < 8; ++r) {
      const int row = mt * 16 + r + half8;
      const int col = t * 16 + n;
      bf2 u = splitf(acc[t][r]);
      if (proj == 0)      { qhi[(size_t)row * H + col] = u.hi; qlo[(size_t)row * H + col] = u.lo; }
      else if (proj == 1) { khi[(size_t)row * H + col] = u.hi; klo[(size_t)row * H + col] = u.lo; }
      else                { vthi[(size_t)col * BS + row] = u.hi; vtlo[(size_t)col * BS + row] = u.lo; }
    }
  }
}

// ---------------------------------------------------------------------------
// Kernel 2: causal flash attention, one wave per 16-query tile, 32-key blocks.
// ---------------------------------------------------------------------------
__global__ __launch_bounds__(256) void attn_kernel(
    const __bf16* __restrict__ qhi, const __bf16* __restrict__ qlo,
    const __bf16* __restrict__ khi, const __bf16* __restrict__ klo,
    const __bf16* __restrict__ vthi, const __bf16* __restrict__ vtlo,
    float* __restrict__ out)
{
  __shared__ __attribute__((aligned(32))) __bf16 pHi[8][16][32];  // per-wave P transpose staging
  __shared__ __attribute__((aligned(32))) __bf16 pLo[8][16][32];

  const int lane = threadIdx.x & 31;
  const int w    = threadIdx.x >> 5;
  const int qt   = blockIdx.x * 8 + w;     // 0..511 query tiles
  const int bat  = qt >> 7;                // batch
  const int lq   = qt & 127;               // local query tile within batch
  const int qbase = bat * S + lq * 16;     // first global query row

  const int  n      = lane & 15;
  const bool hiHalf = lane >= 16;
  const int  akb    = hiHalf ? 16 : 0;
  const int  bkb    = hiHalf ? 16 : 0;
  const int  half8  = hiHalf ? 8  : 0;
  const float SCALE = 11.313708498984761f;  // sqrt(128): reference MULTIPLIES
  const float NEGI  = -__builtin_inff();

  // Preload Q A-fragments: 4 k-steps (head dim 128 = 4*32), hi & lo planes.
  v16bf qa_hi[4], qa_lo[4];
  {
    const size_t qoff = (size_t)(qbase + n) * H;
#pragma unroll
    for (int ks = 0; ks < 4; ++ks) {
      const __bf16* ph = qhi + qoff + ks * 32 + akb;
      const __bf16* pl = qlo + qoff + ks * 32 + akb;
      qa_hi[ks] = cat16(*(const v8bf*)ph, *(const v8bf*)(ph + 8));
      qa_lo[ks] = cat16(*(const v8bf*)pl, *(const v8bf*)(pl + 8));
    }
  }

  v8f o[8] = {};                 // 16x128 output accumulator (8 N-tiles)
  float mrow[8], lrow[8];
#pragma unroll
  for (int r = 0; r < 8; ++r) { mrow[r] = NEGI; lrow[r] = 0.0f; }

  const int nblocks = (lq * 16 + 15) / 32 + 1;   // causal key-block count
  for (int jb = 0; jb < nblocks; ++jb) {
    const int kv0 = jb * 32;

    // Prefetch next key-block's K/V stream (global_prefetch_b8 on gfx1250).
    if (jb + 1 < nblocks) {
      const size_t pk = (size_t)(bat * S + kv0 + 32 + n) * H;
      __builtin_prefetch(khi + pk, 0, 0);
      __builtin_prefetch(klo + pk, 0, 0);
      const size_t pv = (size_t)n * BS + (size_t)(bat * S + kv0 + 32);
      __builtin_prefetch(vthi + pv, 0, 0);
      __builtin_prefetch(vtlo + pv, 0, 0);
    }

    // ---- S = Q K^T : two 16x16 score tiles (keys kv0.., kv0+16..) ----
    v8f c0 = {}, c1 = {};
#pragma unroll
    for (int ks = 0; ks < 4; ++ks) {
      const size_t k0o = (size_t)(bat * S + kv0 + n)      * H + ks * 32 + bkb;
      const size_t k1o = (size_t)(bat * S + kv0 + 16 + n) * H + ks * 32 + bkb;
      v16bf k0h = *(const v16bf*)(khi + k0o);
      v16bf k0l = *(const v16bf*)(klo + k0o);
      v16bf k1h = *(const v16bf*)(khi + k1o);
      v16bf k1l = *(const v16bf*)(klo + k1o);
      c0 = WMMA_BF16(qa_hi[ks], k0h, c0);
      c0 = WMMA_BF16(qa_hi[ks], k0l, c0);
      c0 = WMMA_BF16(qa_lo[ks], k0h, c0);
      c1 = WMMA_BF16(qa_hi[ks], k1h, c1);
      c1 = WMMA_BF16(qa_hi[ks], k1l, c1);
      c1 = WMMA_BF16(qa_lo[ks], k1h, c1);
    }

    // ---- online softmax (reference quirk: kept-but-zero scores -> -inf) ----
    float alpha[8];
#pragma unroll
    for (int r = 0; r < 8; ++r) {
      float s0 = c0[r] * SCALE, s1 = c1[r] * SCALE;
      const int qrow_l = lq * 16 + r + half8;
      if (!((kv0 + n)      <= qrow_l && s0 != 0.0f)) s0 = NEGI;
      if (!((kv0 + 16 + n) <= qrow_l && s1 != 0.0f)) s1 = NEGI;
      float mx = fmaxf(s0, s1);
#pragma unroll
      for (int d = 1; d < 16; d <<= 1) mx = fmaxf(mx, __shfl_xor(mx, d, 32));
      const float mnew = fmaxf(mrow[r], mx);
      const float a  = (mrow[r] == NEGI) ? 0.0f : __expf(mrow[r] - mnew);
      const float p0 = (s0 == NEGI) ? 0.0f : __expf(s0 - mnew);
      const float p1 = (s1 == NEGI) ? 0.0f : __expf(s1 - mnew);
      float ps = p0 + p1;
#pragma unroll
      for (int d = 1; d < 16; d <<= 1) ps += __shfl_xor(ps, d, 32);
      lrow[r] = lrow[r] * a + ps;
      mrow[r] = mnew;
      alpha[r] = a;
      // C-layout -> LDS (row m = r+half8, cols n and n+16), bf16 hi/lo
      const int m = r + half8;
      bf2 u0 = splitf(p0); pHi[w][m][n]      = u0.hi; pLo[w][m][n]      = u0.lo;
      bf2 u1 = splitf(p1); pHi[w][m][16 + n] = u1.hi; pLo[w][m][16 + n] = u1.lo;
    }
#pragma unroll
    for (int t = 0; t < 8; ++t)
#pragma unroll
      for (int r = 0; r < 8; ++r) o[t][r] = o[t][r] * alpha[r];

    __builtin_amdgcn_wave_barrier();   // wave-private LDS: order stores before reads
    v16bf pa_h, pa_l;
    {
      const __bf16* ph = &pHi[w][n][akb];
      const __bf16* pl = &pLo[w][n][akb];
      pa_h = cat16(*(const v8bf*)ph, *(const v8bf*)(ph + 8));
      pa_l = cat16(*(const v8bf*)pl, *(const v8bf*)(pl + 8));
    }
    __builtin_amdgcn_wave_barrier();

    // ---- O += P V : V stored transposed [H][BS] so B-fragments are contiguous ----
    const size_t vrow = (size_t)(bat * S + kv0 + bkb);
#pragma unroll
    for (int t = 0; t < 8; ++t) {
      const size_t vo = (size_t)(t * 16 + n) * BS + vrow;
      v16bf vh = *(const v16bf*)(vthi + vo);
      v16bf vl = *(const v16bf*)(vtlo + vo);
      o[t] = WMMA_BF16(pa_h, vh, o[t]);
      o[t] = WMMA_BF16(pa_h, vl, o[t]);
      o[t] = WMMA_BF16(pa_l, vh, o[t]);
    }
  }

  // ---- epilogue: normalize and store fp32 output (B,S,H) ----
#pragma unroll
  for (int r = 0; r < 8; ++r) {
    const float inv = 1.0f / lrow[r];
    const int row = qbase + r + half8;
#pragma unroll
    for (int t = 0; t < 8; ++t)
      out[(size_t)row * H + t * 16 + n] = o[t][r] * inv;
  }
}

// ---------------------------------------------------------------------------
extern "C" void kernel_launch(void* const* d_in, const int* in_sizes, int n_in,
                              void* d_out, int out_size, void* d_ws, size_t ws_size,
                              hipStream_t stream) {
  const float* x  = (const float*)d_in[0];
  const float* Wq = (const float*)d_in[1];
  const float* Wk = (const float*)d_in[2];
  const float* Wv = (const float*)d_in[3];

  char* ws = (char*)d_ws;
  const size_t plane = (size_t)BS * H * sizeof(__bf16);   // 2 MB
  const size_t wplane = (size_t)3 * H * E * sizeof(__bf16); // 768 KB
  __bf16* qhi  = (__bf16*)(ws + 0 * plane);
  __bf16* qlo  = (__bf16*)(ws + 1 * plane);
  __bf16* khi  = (__bf16*)(ws + 2 * plane);
  __bf16* klo  = (__bf16*)(ws + 3 * plane);
  __bf16* vthi = (__bf16*)(ws + 4 * plane);
  __bf16* vtlo = (__bf16*)(ws + 5 * plane);
  __bf16* whi  = (__bf16*)(ws + 6 * plane);
  __bf16* wlo  = (__bf16*)(ws + 6 * plane + wplane);

  // W conversion: 3*128*1024 elems, 8 per thread -> 192 blocks of 256.
  conv_w_kernel<<<192, 256, 0, stream>>>(Wq, Wk, Wv, whi, wlo);
  // 3 projections * 512 row-tiles = 1536 waves = 192 blocks of 8 waves.
  qkv_proj_kernel<<<192, 256, 0, stream>>>(x, whi, wlo, qhi, qlo, khi, klo, vthi, vtlo);
  // 512 query tiles = 64 blocks of 8 waves.
  attn_kernel<<<64, 256, 0, stream>>>(qhi, qlo, khi, klo, vthi, vtlo, (float*)d_out);
}
